// LOBTransformer_23880018165803
// MI455X (gfx1250) — matrix operand order repaired and checked by reading, not compiled, not run on hardware
//
#include <hip/hip_runtime.h>
#include <math.h>

#define L_  4
#define B_  4
#define S_  1024
#define D_  512
#define H_  8
#define FF_ 2048
#define V_  10000
#define DK_ 64

typedef __attribute__((ext_vector_type(16))) __bf16 v16bf;
typedef __attribute__((ext_vector_type(8)))  float  v8f;

union FragU { uint4 u[2]; v16bf v; };

__device__ __forceinline__ unsigned short f2bf(float f) {
  unsigned u = __float_as_uint(f);
  u += 0x7FFFu + ((u >> 16) & 1u);          // round-to-nearest-even
  return (unsigned short)(u >> 16);
}

__device__ __forceinline__ uint4 pack8(const unsigned short* t) {
  uint4 u;
  u.x = (unsigned)t[0] | ((unsigned)t[1] << 16);
  u.y = (unsigned)t[2] | ((unsigned)t[3] << 16);
  u.z = (unsigned)t[4] | ((unsigned)t[5] << 16);
  u.w = (unsigned)t[6] | ((unsigned)t[7] << 16);
  return u;
}

// ---------------------------------------------------------------------------
// Embedding gather: x[b,s,:] = tok[id] + pos[s] + time[tb]
// ---------------------------------------------------------------------------
__global__ __launch_bounds__(128) void embed_kernel(
    const int* __restrict__ ids, const int* __restrict__ tbk,
    const float* __restrict__ tok, const float* __restrict__ pos,
    const float* __restrict__ tim, float* __restrict__ x)
{
  int r  = blockIdx.x;            // b*S + s
  int s  = r & (S_ - 1);
  int id = ids[r];
  int tb = tbk[r];
  for (int d = threadIdx.x; d < D_; d += 128)
    x[(size_t)r * D_ + d] = tok[(size_t)id * D_ + d] + pos[(size_t)s * D_ + d]
                          + tim[(size_t)tb * D_ + d];
}

// ---------------------------------------------------------------------------
// LayerNorm: 8 waves per block, one wave32 per row; writes bf16
// ---------------------------------------------------------------------------
__global__ __launch_bounds__(256) void ln_kernel(
    const float* __restrict__ x, const float* __restrict__ g,
    const float* __restrict__ b, unsigned short* __restrict__ out)
{
  int wave = threadIdx.x >> 5, lane = threadIdx.x & 31;
  int row = blockIdx.x * 8 + wave;
  const float* xr = x + (size_t)row * D_;
  float v[16], s = 0.f;
  for (int i = 0; i < 16; ++i) { v[i] = xr[lane + i * 32]; s += v[i]; }
  for (int m = 1; m < 32; m <<= 1) s += __shfl_xor(s, m, 32);
  float mean = s * (1.f / D_);
  float q = 0.f;
  for (int i = 0; i < 16; ++i) { float d = v[i] - mean; q += d * d; }
  for (int m = 1; m < 32; m <<= 1) q += __shfl_xor(q, m, 32);
  float rstd = rsqrtf(q * (1.f / D_) + 1e-5f);
  for (int i = 0; i < 16; ++i) {
    int d = lane + i * 32;
    out[(size_t)row * D_ + d] = f2bf((v[i] - mean) * rstd * g[d] + b[d]);
  }
}

// ---------------------------------------------------------------------------
// Tiled bf16 WMMA GEMM: C(MxN) = A(MxK,bf16) * B(KxN,f32->bf16)
// 256 threads = 8 waves; block tile 128x128; each wave 32x64 (2x4 WMMA frags)
// ---------------------------------------------------------------------------
#define MODE_F32      0   // plain f32 store (logits)
#define MODE_F32_RES  1   // out = resid + acc + bias   (f32)
#define MODE_BF16     2   // bf16 store (+bias)
#define MODE_GELU     3   // bf16 store of gelu(acc+bias)
#define MODE_QKV      4   // bf16 store scattered to (B,H,S,DK)

#define TM 128
#define TN 128
#define TK 32
#define APAD 40
#define BPAD 40

__global__ __launch_bounds__(256) void gemm_kernel(
    const unsigned short* __restrict__ A, const float* __restrict__ Bw,
    const float* __restrict__ bias, const float* __restrict__ resid,
    void* __restrict__ outp, int M, int N, int K, int bTrans, int mode)
{
  __shared__ alignas(16) unsigned short As[TM * APAD];
  __shared__ alignas(16) unsigned short Bs[TN * BPAD];

  int tid   = threadIdx.x;
  int mbase = blockIdx.y * TM;
  int nbase = blockIdx.x * TN;
  int wave  = tid >> 5, lane = tid & 31;
  int half  = lane >> 4, l16 = lane & 15;
  int wm = (wave & 3) * 32, wn = (wave >> 2) * 64;
  bool full = (nbase + TN <= N);          // uniform: only last logits column is partial

  v8f acc[2][4];
  for (int mi = 0; mi < 2; ++mi)
    for (int ni = 0; ni < 4; ++ni)
      for (int r = 0; r < 8; ++r) acc[mi][ni][r] = 0.f;

  for (int kb = 0; kb < K; kb += TK) {
    __syncthreads();
    // stage A tile (bf16, row-major, K-contiguous): 2 x b128 per thread
    for (int it = 0; it < 2; ++it) {
      int slot = tid + it * 256;
      int row = slot >> 2, c = slot & 3;
      uint4 d = *(const uint4*)(A + (size_t)(mbase + row) * K + kb + c * 8);
      *(uint4*)(&As[row * APAD + c * 8]) = d;
    }
    // stage B tile into [n][k] bf16; 16 elements/thread, 2 packed b128 stores
    {
      unsigned short t[16];
      int n, k0;
      if (!bTrans) {
        n = tid & 127; k0 = (tid >> 7) * 16;
        int gn = nbase + n;
        if (full) {
          for (int i = 0; i < 16; ++i)
            t[i] = f2bf(Bw[(size_t)(kb + k0 + i) * N + gn]);
        } else {
          for (int i = 0; i < 16; ++i)
            t[i] = (gn < N) ? f2bf(Bw[(size_t)(kb + k0 + i) * N + gn]) : 0;
        }
      } else {          // B[k][n] = Bw[n][k]  (tok_emb^T for logits)
        n = tid >> 1; k0 = (tid & 1) * 16;
        int gn = nbase + n;
        if (full || gn < N) {
          const float4* src = (const float4*)(Bw + (size_t)gn * K + kb + k0);
          float4 f0 = src[0], f1 = src[1], f2 = src[2], f3 = src[3];
          t[0]=f2bf(f0.x); t[1]=f2bf(f0.y); t[2]=f2bf(f0.z); t[3]=f2bf(f0.w);
          t[4]=f2bf(f1.x); t[5]=f2bf(f1.y); t[6]=f2bf(f1.z); t[7]=f2bf(f1.w);
          t[8]=f2bf(f2.x); t[9]=f2bf(f2.y); t[10]=f2bf(f2.z); t[11]=f2bf(f2.w);
          t[12]=f2bf(f3.x); t[13]=f2bf(f3.y); t[14]=f2bf(f3.z); t[15]=f2bf(f3.w);
        } else {
          for (int i = 0; i < 16; ++i) t[i] = 0;
        }
      }
      *(uint4*)(&Bs[n * BPAD + k0])     = pack8(&t[0]);
      *(uint4*)(&Bs[n * BPAD + k0 + 8]) = pack8(&t[8]);
    }
    __syncthreads();
    if (kb + TK < K)    // global_prefetch_b8 of next A tile
      __builtin_prefetch(A + (size_t)(mbase + (tid & 127)) * K + kb + TK, 0, 1);

    FragU a[2], b[4];
    for (int mi = 0; mi < 2; ++mi) {        // A: interleaved K pattern
      const unsigned short* ar = &As[(wm + mi * 16 + l16) * APAD];
      a[mi].u[0] = *(const uint4*)(ar + 8 * half);
      a[mi].u[1] = *(const uint4*)(ar + 16 + 8 * half);
    }
    for (int ni = 0; ni < 4; ++ni) {        // B: interleaved K pattern
      const unsigned short* br = &Bs[(wn + ni * 16 + l16) * BPAD];
      b[ni].u[0] = *(const uint4*)(br + 8 * half);
      b[ni].u[1] = *(const uint4*)(br + 16 + 8 * half);
    }
    for (int mi = 0; mi < 2; ++mi)
      for (int ni = 0; ni < 4; ++ni)
        acc[mi][ni] = __builtin_amdgcn_wmma_f32_16x16x32_bf16(
            false, a[mi].v, false, b[ni].v, (short)0, acc[mi][ni], false, false);
  }

  // epilogue
  for (int mi = 0; mi < 2; ++mi)
    for (int ni = 0; ni < 4; ++ni)
      for (int r = 0; r < 8; ++r) {
        int m = mbase + wm + mi * 16 + r + 8 * half;
        int n = nbase + wn + ni * 16 + l16;
        if (n >= N) continue;
        float v = acc[mi][ni][r];
        if (mode == MODE_F32) {
          ((float*)outp)[(size_t)m * N + n] = v;
        } else if (mode == MODE_F32_RES) {
          ((float*)outp)[(size_t)m * N + n] =
              resid[(size_t)m * N + n] + v + (bias ? bias[n] : 0.f);
        } else if (mode == MODE_BF16) {
          ((unsigned short*)outp)[(size_t)m * N + n] =
              f2bf(v + (bias ? bias[n] : 0.f));
        } else if (mode == MODE_GELU) {
          float t = v + bias[n];
          t = 0.5f * t * (1.f + erff(t * 0.70710678118f));
          ((unsigned short*)outp)[(size_t)m * N + n] = f2bf(t);
        } else {        // MODE_QKV: scatter to (B,H,S,DK) bf16
          int s = m & (S_ - 1), bb = m >> 10;
          int hh = n >> 6, dk = n & (DK_ - 1);
          ((unsigned short*)outp)[(((size_t)bb * H_ + hh) * S_ + s) * DK_ + dk] =
              f2bf(v);
        }
      }
}

// ---------------------------------------------------------------------------
// Flash attention: 4 waves/block, each wave owns 16 q rows; WMMA for QK^T & PV
// q,k,v bf16 in (B,H,S,DK); output bf16 directly in (B,S,D) layout
// ---------------------------------------------------------------------------
__global__ __launch_bounds__(128) void attn_kernel(
    const unsigned short* __restrict__ Q, const unsigned short* __restrict__ Kg,
    const unsigned short* __restrict__ Vg, const int* __restrict__ amask,
    unsigned short* __restrict__ Obsd)
{
  __shared__ alignas(16) unsigned short Vt[64 * 40];       // [dk][kk]
  __shared__ alignas(16) unsigned short Pl[4][16 * 40];    // per-wave [q][kk]

  int tid = threadIdx.x, wave = tid >> 5, lane = tid & 31;
  int half = lane >> 4, l16 = lane & 15;
  int bh = blockIdx.y, b = bh >> 3, h = bh & 7;
  int qblock = blockIdx.x * 64;
  int qbase  = qblock + wave * 16;

  const unsigned short* Qp = Q  + (size_t)bh * S_ * DK_;
  const unsigned short* Kp = Kg + (size_t)bh * S_ * DK_;
  const unsigned short* Vp = Vg + (size_t)bh * S_ * DK_;

  // Q A-fragments (dk halves 0-31, 32-63), interleaved-K pattern
  FragU qa[2];
  {
    const unsigned short* qr = Qp + (size_t)(qbase + l16) * DK_;
    qa[0].u[0] = *(const uint4*)(qr + 8 * half);
    qa[0].u[1] = *(const uint4*)(qr + 16 + 8 * half);
    qa[1].u[0] = *(const uint4*)(qr + 32 + 8 * half);
    qa[1].u[1] = *(const uint4*)(qr + 48 + 8 * half);
  }

  float mrow[8], lrow[8];
  v8f Of[4];
  for (int r = 0; r < 8; ++r) { mrow[r] = -1e30f; lrow[r] = 0.f; }
  for (int t = 0; t < 4; ++t)
    for (int r = 0; r < 8; ++r) Of[t][r] = 0.f;

  const float scale = 0.125f;                 // 1/sqrt(64)
  int kbn = (qblock + 64) >> 5;               // causal extent in 32-key blocks

  for (int kb = 0; kb < kbn; ++kb) {
    __syncthreads();
    // stage V tile transposed: Vt[dk][kk]
    {
      int kk = tid & 31, dkc = tid >> 5;
      const unsigned short* vr = Vp + (size_t)(kb * 32 + kk) * DK_ + dkc * 16;
      unsigned short tmp[16];
      *(uint4*)&tmp[0] = *(const uint4*)(vr);
      *(uint4*)&tmp[8] = *(const uint4*)(vr + 8);
      for (int i = 0; i < 16; ++i) Vt[(dkc * 16 + i) * 40 + kk] = tmp[i];
    }
    __syncthreads();

    // scores: 2 key sub-tiles x 2 dk fragments = 4 WMMAs
    v8f sc[2];
    int kcol[2]; int amv[2];
    for (int t2 = 0; t2 < 2; ++t2) {
      for (int r = 0; r < 8; ++r) sc[t2][r] = 0.f;
      kcol[t2] = kb * 32 + t2 * 16 + l16;
      amv[t2]  = amask[b * S_ + kcol[t2]];
      const unsigned short* kr = Kp + (size_t)kcol[t2] * DK_;
      FragU k0, k1;                           // B frags: interleaved K pattern
      k0.u[0] = *(const uint4*)(kr + 8 * half);
      k0.u[1] = *(const uint4*)(kr + 16 + 8 * half);
      k1.u[0] = *(const uint4*)(kr + 32 + 8 * half);
      k1.u[1] = *(const uint4*)(kr + 48 + 8 * half);
      sc[t2] = __builtin_amdgcn_wmma_f32_16x16x32_bf16(
          false, qa[0].v, false, k0.v, (short)0, sc[t2], false, false);
      sc[t2] = __builtin_amdgcn_wmma_f32_16x16x32_bf16(
          false, qa[1].v, false, k1.v, (short)0, sc[t2], false, false);
    }

    // online softmax over the 32 new keys (row reductions across 16 lanes)
    float alpha[8];
    for (int r = 0; r < 8; ++r) {
      int qg = qbase + r + 8 * half;
      float v0 = sc[0][r] * scale;
      float v1 = sc[1][r] * scale;
      if (kcol[0] > qg || amv[0] == 0) v0 = -1e30f;
      if (kcol[1] > qg || amv[1] == 0) v1 = -1e30f;
      float mx = fmaxf(v0, v1);
      for (int m = 1; m < 16; m <<= 1) mx = fmaxf(mx, __shfl_xor(mx, m, 32));
      float mn = fmaxf(mrow[r], mx);
      float p0 = __expf(v0 - mn), p1 = __expf(v1 - mn);
      float sm = p0 + p1;
      for (int m = 1; m < 16; m <<= 1) sm += __shfl_xor(sm, m, 32);
      float al = __expf(mrow[r] - mn);
      lrow[r] = lrow[r] * al + sm;
      mrow[r] = mn;
      alpha[r] = al;
      Pl[wave][(r + 8 * half) * 40 + l16]      = f2bf(p0);
      Pl[wave][(r + 8 * half) * 40 + 16 + l16] = f2bf(p1);
    }
    for (int t = 0; t < 4; ++t)
      for (int r = 0; r < 8; ++r) Of[t][r] *= alpha[r];
    __syncthreads();

    // PV: P(16x32) A-frag from LDS, 4 dk-tiles of V as B-frags
    FragU pa;
    const unsigned short* pr = &Pl[wave][l16 * 40];
    pa.u[0] = *(const uint4*)(pr + 8 * half);
    pa.u[1] = *(const uint4*)(pr + 16 + 8 * half);
    for (int t3 = 0; t3 < 4; ++t3) {
      FragU vb;
      const unsigned short* vr = &Vt[(t3 * 16 + l16) * 40];
      vb.u[0] = *(const uint4*)(vr + 8 * half);
      vb.u[1] = *(const uint4*)(vr + 16 + 8 * half);
      Of[t3] = __builtin_amdgcn_wmma_f32_16x16x32_bf16(
          false, pa.v, false, vb.v, (short)0, Of[t3], false, false);
    }
  }

  // normalize and write O directly in (B,S,D) bf16 layout
  for (int t3 = 0; t3 < 4; ++t3)
    for (int r = 0; r < 8; ++r) {
      int srow = qbase + r + 8 * half;
      float v = Of[t3][r] / lrow[r];
      Obsd[((size_t)b * S_ + srow) * D_ + h * DK_ + t3 * 16 + l16] = f2bf(v);
    }
}

// ---------------------------------------------------------------------------
extern "C" void kernel_launch(void* const* d_in, const int* in_sizes, int n_in,
                              void* d_out, int out_size, void* d_ws, size_t ws_size,
                              hipStream_t stream) {
  (void)in_sizes; (void)n_in; (void)out_size; (void)ws_size;
  const int*   ids  = (const int*)  d_in[0];
  const int*   tbk  = (const int*)  d_in[1];
  const int*   amsk = (const int*)  d_in[2];
  const float* tok  = (const float*)d_in[3];
  const float* pos  = (const float*)d_in[4];
  const float* tim  = (const float*)d_in[5];
  const float* Wq   = (const float*)d_in[6];
  const float* Wk   = (const float*)d_in[7];
  const float* Wv   = (const float*)d_in[8];
  const float* Wo   = (const float*)d_in[9];
  const float* bo   = (const float*)d_in[10];
  const float* ln1g = (const float*)d_in[11];
  const float* ln1b = (const float*)d_in[12];
  const float* W1   = (const float*)d_in[13];
  const float* b1   = (const float*)d_in[14];
  const float* W2   = (const float*)d_in[15];
  const float* b2   = (const float*)d_in[16];
  const float* ln2g = (const float*)d_in[17];
  const float* ln2b = (const float*)d_in[18];
  const float* lnfg = (const float*)d_in[19];
  const float* lnfb = (const float*)d_in[20];

  char* ws = (char*)d_ws;
  const size_t MB = (size_t)1 << 20;
  float*          x   = (float*)(ws);                     // 8 MB
  unsigned short* hb  = (unsigned short*)(ws +  8 * MB);  // 4 MB
  unsigned short* qb  = (unsigned short*)(ws + 12 * MB);  // 4 MB
  unsigned short* kb  = (unsigned short*)(ws + 16 * MB);  // 4 MB
  unsigned short* vb  = (unsigned short*)(ws + 20 * MB);  // 4 MB
  unsigned short* ob  = (unsigned short*)(ws + 24 * MB);  // 4 MB
  unsigned short* mid = (unsigned short*)(ws + 28 * MB);  // 16 MB

  const int M = B_ * S_;
  dim3 gD(D_ / TN, M / TM);                 // 4 x 32
  dim3 gF(FF_ / TN, M / TM);                // 16 x 32
  dim3 gV((V_ + TN - 1) / TN, M / TM);      // 79 x 32

  embed_kernel<<<dim3(M), dim3(128), 0, stream>>>(ids, tbk, tok, pos, tim, x);

  for (int l = 0; l < L_; ++l) {
    ln_kernel<<<M / 8, 256, 0, stream>>>(x, ln1g + l * D_, ln1b + l * D_, hb);
    gemm_kernel<<<gD, 256, 0, stream>>>(hb, Wq + (size_t)l * D_ * D_, nullptr,
                                        nullptr, qb, M, D_, D_, 0, MODE_QKV);
    gemm_kernel<<<gD, 256, 0, stream>>>(hb, Wk + (size_t)l * D_ * D_, nullptr,
                                        nullptr, kb, M, D_, D_, 0, MODE_QKV);
    gemm_kernel<<<gD, 256, 0, stream>>>(hb, Wv + (size_t)l * D_ * D_, nullptr,
                                        nullptr, vb, M, D_, D_, 0, MODE_QKV);
    attn_kernel<<<dim3(S_ / 64, B_ * H_), 128, 0, stream>>>(qb, kb, vb, amsk, ob);
    gemm_kernel<<<gD, 256, 0, stream>>>(ob, Wo + (size_t)l * D_ * D_, bo + l * D_,
                                        x, x, M, D_, D_, 0, MODE_F32_RES);
    ln_kernel<<<M / 8, 256, 0, stream>>>(x, ln2g + l * D_, ln2b + l * D_, hb);
    gemm_kernel<<<gF, 256, 0, stream>>>(hb, W1 + (size_t)l * D_ * FF_, b1 + l * FF_,
                                        nullptr, mid, M, FF_, D_, 0, MODE_GELU);
    gemm_kernel<<<gD, 256, 0, stream>>>(mid, W2 + (size_t)l * FF_ * D_, b2 + l * D_,
                                        x, x, M, D_, FF_, 0, MODE_F32_RES);
  }

  ln_kernel<<<M / 8, 256, 0, stream>>>(x, lnfg, lnfb, hb);
  gemm_kernel<<<gV, 256, 0, stream>>>(hb, tok, nullptr, nullptr,
                                      (float*)d_out, M, V_, D_, 1, MODE_F32);
}
